// MomentVideo_84430467104784
// MI455X (gfx1250) — matrix operand order repaired and checked by reading, not compiled, not run on hardware
//
#include <hip/hip_runtime.h>
#include <hip/hip_bf16.h>
#include <math.h>

// ---------------------------------------------------------------------------
// Problem constants (from the reference)
// ---------------------------------------------------------------------------
#define BATCH 32
#define TV    1024
#define TS    64
#define DIN   1024
#define DDIM  512
#define EPSV  1e-15f

typedef __attribute__((ext_vector_type(16))) __bf16 bf16x16;
typedef __attribute__((ext_vector_type(8)))  __bf16 bf16x8;
typedef __attribute__((ext_vector_type(8)))  float  f32x8;
typedef __attribute__((ext_vector_type(4)))  float  f32x4;
typedef __attribute__((ext_vector_type(4)))  unsigned int u32x4;
typedef __attribute__((ext_vector_type(8)))  int    i32x8;
typedef __attribute__((ext_vector_type(4)))  int    i32x4;

// WMMA wrapper: D = A(16x32 bf16) * B(32x16 bf16) + C(16x16 f32)
__device__ __forceinline__ f32x8 wmma_bf16(bf16x16 a, bf16x16 b, f32x8 c) {
    return __builtin_amdgcn_wmma_f32_16x16x32_bf16(false, a, false, b,
                                                   (short)0, c, false, false);
}

// A-fragment (16x32, row r = lane&15, half h = lane>>4) from a row-major
// bf16 row pointer at the K-chunk base. K order used here (contiguous, matched by B):
//   element e -> K = 16*h + e
__device__ __forceinline__ bf16x16 load_a_frag(const __bf16* rowp, int h) {
    bf16x8 lo = *(const bf16x8*)(rowp + 16 * h);
    bf16x8 hi = *(const bf16x8*)(rowp + 16 * h + 8);
    bf16x16 a;
#pragma unroll
    for (int e = 0; e < 8; ++e) { a[e] = lo[e]; a[e + 8] = hi[e]; }
    return a;
}

// B-fragment (32x16, col n = lane&15, half h = lane>>4) from a pointer to 16
// contiguous bf16 K-values for this lane's column (pointer at K = 16h).
__device__ __forceinline__ bf16x16 load_b_frag(const __bf16* colp) {
    bf16x8 lo = *(const bf16x8*)(colp);
    bf16x8 hi = *(const bf16x8*)(colp + 8);
    bf16x16 b;
#pragma unroll
    for (int e = 0; e < 8; ++e) { b[e] = lo[e]; b[e + 8] = hi[e]; }
    return b;
}

// ---------------------------------------------------------------------------
// Tensor Data Mover: 2D tiled load S^T[512][64] bf16 -> LDS with pitch 72
// halves (pad 8 halves = 4 DWORDs after every row of 64 halves = 32 DWORDs).
// D# per CDNA5 ISA.
// ---------------------------------------------------------------------------
__device__ __forceinline__ void tdm_load_St(const __bf16* src, unsigned lds_byte) {
    unsigned long long ga = (unsigned long long)(uintptr_t)src;
    u32x4 g0;
    g0[0] = 1u;                                        // count=1, user mode
    g0[1] = lds_byte;                                  // lds_addr
    g0[2] = (unsigned)(ga & 0xffffffffu);              // global_addr[31:0]
    g0[3] = (unsigned)((ga >> 32) & 0x01ffffffu)       // global_addr[56:32]
          | (2u << 30);                                // type = 2 ("image")
    i32x8 g1;
    g1[0] = (int)((1u << 16)      // data_size = 1 -> 2 bytes
                | (1u << 20)      // pad_enable
                | (4u << 22)      // pad_interval: 4 -> every 32 DWORDs (128B)
                | (3u << 25));    // pad_amount:   3 -> 4 DWORDs (16B)
    g1[1] = (int)(64u << 16);     // tensor_dim0 = 64  (bits[79:48] low half)
    g1[2] = (int)(512u << 16);    // tensor_dim1 = 512 (bits[111:80] low half)
    g1[3] = (int)(64u << 16);     // tile_dim0 = 64    (bits[127:112])
    g1[4] = (int)512;             // tile_dim1 = 512   (bits[143:128])
    g1[5] = (int)64;              // tensor_dim0_stride = 64
    g1[6] = 0;
    g1[7] = 0;
    i32x4 z4 = {0, 0, 0, 0};
#if defined(__clang_major__) && __clang_major__ >= 23
    i32x8 z8 = {0, 0, 0, 0, 0, 0, 0, 0};
    __builtin_amdgcn_tensor_load_to_lds(g0, g1, z4, z4, z8, 0);
#else
    __builtin_amdgcn_tensor_load_to_lds(g0, g1, z4, z4, 0);
#endif
    __builtin_amdgcn_s_wait_tensorcnt((short)0);
}

// ---------------------------------------------------------------------------
// Kernel 0a: W[K x N] f32 -> Wt[N x K] bf16  (weights are tiny, L2 resident)
// ---------------------------------------------------------------------------
__global__ void transpose_w_bf16(const float* __restrict__ W,
                                 __bf16* __restrict__ Wt, int K, int N) {
    int idx = blockIdx.x * 256 + threadIdx.x;        // over N*K, k fastest
    if (idx >= N * K) return;
    int n = idx / K;
    int k = idx - n * K;
    Wt[idx] = (__bf16)W[(size_t)k * N + n];
}

// ---------------------------------------------------------------------------
// Kernel 0b: srep[b][k][d] -> srepT[b][d][k]   (bf16, 1M elements)
// ---------------------------------------------------------------------------
__global__ void transpose_srep(const __bf16* __restrict__ srep,
                               __bf16* __restrict__ srepT) {
    int idx = blockIdx.x * 256 + threadIdx.x;        // k fastest
    int k = idx & (TS - 1);
    int d = (idx >> 6) & (DDIM - 1);
    int b = idx >> 15;
    srepT[idx] = srep[((size_t)b * TS + k) * DDIM + d];
}

// ---------------------------------------------------------------------------
// Kernel 1: encoder GEMM  Crep[M x N](bf16) = A[M x K](f32) * Wt^T + bias
//   Wt is the pre-transposed bf16 weight [N x K], so BOTH operand fragments
//   read contiguous-K rows -> all LDS staging is coalesced ds_store_b128.
// Workgroup tile 64x64, 8 waves in a 4(M) x 2(N) grid; per-wave 16x32 output.
// ---------------------------------------------------------------------------
#define GEMM_BK 32
#define LDS_PITCH 40   // 32 + 8 bf16 pad -> 80B row stride, conflict-free

__global__ void encode_gemm(const float* __restrict__ A,
                            const __bf16* __restrict__ Wt,
                            const float* __restrict__ bias,
                            __bf16* __restrict__ Crep,
                            int M, int K, int N) {
    __shared__ __bf16 As[64 * LDS_PITCH];           // [m][k]
    __shared__ __bf16 Bs[64 * LDS_PITCH];           // [n][k]

    const int tid  = threadIdx.x;
    const int wid  = tid >> 5;
    const int lane = tid & 31;
    const int h    = lane >> 4;
    const int ln   = lane & 15;

    const int wm = wid & 3;        // 0..3  -> M sub-tile (16 rows)
    const int wn = wid >> 2;       // 0..1  -> N sub-tile (32 cols)

    const int n0 = blockIdx.x * 64;
    const int m0 = blockIdx.y * 64;

    f32x8 acc0 = {0.f, 0.f, 0.f, 0.f, 0.f, 0.f, 0.f, 0.f};
    f32x8 acc1 = acc0;

    const int a_row = tid >> 2;            // 0..63
    const int a_c0  = (tid & 3) * 8;       // 0,8,16,24
    const int b_row = tid >> 2;            // 0..63 (n)
    const int b_c0  = (tid & 3) * 8;       // 0,8,16,24 (k halves)

    for (int kb = 0; kb < K; kb += GEMM_BK) {
        // A tile: 64 rows x 32 K, f32 -> bf16 (stores merge to ds_store_b128)
        {
            const float* src = A + (size_t)(m0 + a_row) * K + kb + a_c0;
            f32x4 v0 = *(const f32x4*)(src);
            f32x4 v1 = *(const f32x4*)(src + 4);
            __bf16* dst = As + a_row * LDS_PITCH + a_c0;
#pragma unroll
            for (int e = 0; e < 4; ++e) { dst[e] = (__bf16)v0[e]; dst[4 + e] = (__bf16)v1[e]; }
        }
        // B tile: 64 n-rows x 32 K halves, straight bf16 row-major copy
        {
            bf16x8 w = *(const bf16x8*)(Wt + (size_t)(n0 + b_row) * K + kb + b_c0);
            *(bf16x8*)(Bs + b_row * LDS_PITCH + b_c0) = w;
        }
        __syncthreads();

        bf16x16 afrag = load_a_frag(As + (wm * 16 + ln) * LDS_PITCH, h);
        bf16x16 b0 = load_b_frag(Bs + (wn * 32 + ln) * LDS_PITCH + 16 * h);
        bf16x16 b1 = load_b_frag(Bs + (wn * 32 + 16 + ln) * LDS_PITCH + 16 * h);
        acc0 = wmma_bf16(afrag, b0, acc0);
        acc1 = wmma_bf16(afrag, b1, acc1);
        __syncthreads();
    }

    // Epilogue: bias add, convert to bf16 (C/D layout: row = v + 8h)
#pragma unroll
    for (int v = 0; v < 8; ++v) {
        const int row  = m0 + wm * 16 + 8 * h + v;
        const int col0 = n0 + wn * 32 + ln;
        const int col1 = col0 + 16;
        Crep[(size_t)row * N + col0] = (__bf16)(acc0[v] + bias[col0]);
        Crep[(size_t)row * N + col1] = (__bf16)(acc1[v] + bias[col1]);
    }
}

// ---------------------------------------------------------------------------
// Kernel 2: row L2 norms of a [R x 512] bf16 matrix, one wave per row.
// ---------------------------------------------------------------------------
__global__ void row_norms(const __bf16* __restrict__ rep, float* __restrict__ nrm, int R) {
    const int wid  = threadIdx.x >> 5;
    const int lane = threadIdx.x & 31;
    const int row  = blockIdx.x * 8 + wid;
    if (row >= R) return;
    const __bf16* p = rep + (size_t)row * DDIM + lane * 16;
    bf16x8 u0 = *(const bf16x8*)(p);
    bf16x8 u1 = *(const bf16x8*)(p + 8);
    float s = 0.f;
#pragma unroll
    for (int e = 0; e < 8; ++e) {
        float f0 = (float)u0[e], f1 = (float)u1[e];
        s += f0 * f0 + f1 * f1;
    }
#pragma unroll
    for (int m = 16; m > 0; m >>= 1) s += __shfl_xor(s, m, 32);
    if (lane == 0) nrm[row] = sqrtf(s);
}

// ---------------------------------------------------------------------------
// Kernel 3: one workgroup per (video i, sentence j) pair.
//   C = V S^T / (nv ns + eps); A = softmax_rows(C); att = A S;
//   score = <v,att>/(|v||att|+eps); out[i][j] = logsumexp(score)
// S^T arrives in LDS via the Tensor Data Mover (padded pitch from the D#).
// ---------------------------------------------------------------------------
#define ST_PITCH 72     // 64 + 8 bf16 pad (144B row stride, conflict-free)
#define ASCR_PITCH 72
#define SMEM_ST    (DDIM * ST_PITCH * 2)           // 73728 B  : S^T  [d][k]
#define SMEM_ASCR  (8 * 16 * ASCR_PITCH * 2)       // 18432 B  : per-wave A tiles
#define SMEM_SCOR  (TV * 4)                        //  4096 B  : per-frame scores
#define SMEM_RED   (256 * 4)                       //  1024 B  : block reduction
#define SMEM_PAIR  (SMEM_ST + SMEM_ASCR + SMEM_SCOR + SMEM_RED)

__global__ void pair_kernel(const __bf16* __restrict__ vrep,
                            const __bf16* __restrict__ srep,
                            const __bf16* __restrict__ srepT,
                            const float* __restrict__ nv_g,
                            const float* __restrict__ ns_g,
                            float* __restrict__ out) {
    extern __shared__ char smem[];
    __bf16* St     = (__bf16*)smem;                         // [512][ST_PITCH]
    __bf16* Ascr   = St + DDIM * ST_PITCH;                  // per-wave [16][ASCR_PITCH]
    float*  scores = (float*)(Ascr + 8 * 16 * ASCR_PITCH);  // [1024]
    float*  red    = scores + TV;                           // [256]

    const int j = blockIdx.x;   // sentence batch index
    const int i = blockIdx.y;   // video batch index

    const int tid  = threadIdx.x;
    const int wid  = tid >> 5;
    const int lane = tid & 31;
    const int h    = lane >> 4;
    const int ln   = lane & 15;

    const __bf16* Vi = vrep + (size_t)i * TV * DDIM;
    const __bf16* Sj = srep + (size_t)j * TS * DDIM;
    const float*  nv = nv_g + (size_t)i * TV;
    const float*  ns = ns_g + (size_t)j * TS;

    // TDM: one wave DMA-loads S_j^T [512][64] bf16 into padded-pitch LDS.
    if (wid == 0) {
        tdm_load_St(srepT + (size_t)j * DDIM * TS, (unsigned)(uintptr_t)St);
    }
    __syncthreads();

    __bf16* Aw = Ascr + wid * 16 * ASCR_PITCH;

    // Each wave owns 8 m-tiles of 16 video frames (64 tiles total).
    for (int mt = wid; mt < TV / 16; mt += 8) {
        const int mbase = mt * 16;

        // ---- Stage 1: C tile [16 x 64] = V[16 x 512] * S^T --------------
        f32x8 acc[4];
#pragma unroll
        for (int nt = 0; nt < 4; ++nt)
#pragma unroll
            for (int v = 0; v < 8; ++v) acc[nt][v] = 0.f;

        for (int kb = 0; kb < DDIM; kb += 32) {
            bf16x16 afrag = load_a_frag(Vi + (size_t)(mbase + ln) * DDIM + kb, h);
#pragma unroll
            for (int nt = 0; nt < 4; ++nt) {
                bf16x16 bfrag = load_b_frag(Sj + (size_t)(nt * 16 + ln) * DDIM + kb + 16 * h);
                acc[nt] = wmma_bf16(afrag, bfrag, acc[nt]);
            }
        }

        // ---- scale by 1/(nv*ns+eps), softmax over 64 cols per row -------
        float nvrow[8];
#pragma unroll
        for (int v = 0; v < 8; ++v) nvrow[v] = nv[mbase + 8 * h + v];
#pragma unroll
        for (int nt = 0; nt < 4; ++nt) {
            const float nsv = ns[nt * 16 + ln];
#pragma unroll
            for (int v = 0; v < 8; ++v)
                acc[nt][v] = acc[nt][v] / (nvrow[v] * nsv + EPSV);
        }
#pragma unroll
        for (int v = 0; v < 8; ++v) {
            float m = fmaxf(fmaxf(acc[0][v], acc[1][v]), fmaxf(acc[2][v], acc[3][v]));
#pragma unroll
            for (int msk = 8; msk > 0; msk >>= 1) m = fmaxf(m, __shfl_xor(m, msk, 32));
            float s = 0.f;
#pragma unroll
            for (int nt = 0; nt < 4; ++nt) {
                float e = __expf(acc[nt][v] - m);
                acc[nt][v] = e;
                s += e;
            }
#pragma unroll
            for (int msk = 8; msk > 0; msk >>= 1) s += __shfl_xor(s, msk, 32);
            const float inv = 1.f / s;
            // A tile (C/D layout: row = v+8h, col = nt*16+ln) -> wave scratch
#pragma unroll
            for (int nt = 0; nt < 4; ++nt)
                Aw[(v + 8 * h) * ASCR_PITCH + nt * 16 + ln] = (__bf16)(acc[nt][v] * inv);
        }
        // wave-private LDS scratch: same-wave DS ops are in-order; no barrier

        // ---- Stage 2: att = A[16x64] * S[64x512], fused reductions ------
        float num_[8], sq_[8];
#pragma unroll
        for (int v = 0; v < 8; ++v) { num_[v] = 0.f; sq_[v] = 0.f; }

        for (int nt2 = 0; nt2 < DDIM / 16; ++nt2) {
            const int dbase = nt2 * 16;
            f32x8 attacc;
#pragma unroll
            for (int v = 0; v < 8; ++v) attacc[v] = 0.f;
#pragma unroll
            for (int kk = 0; kk < 2; ++kk) {          // K = 64 sentence tokens
                const int kb2 = kk * 32;
                bf16x16 afrag = load_a_frag(Aw + ln * ASCR_PITCH + kb2, h);
                bf16x16 bfrag = load_b_frag(St + (dbase + ln) * ST_PITCH + kb2 + 16 * h);
                attacc = wmma_bf16(afrag, bfrag, attacc);
            }
#pragma unroll
            for (int v = 0; v < 8; ++v) {
                const float a  = attacc[v];
                const float vv = (float)Vi[(size_t)(mbase + 8 * h + v) * DDIM + dbase + ln];
                num_[v] += vv * a;
                sq_[v]  += a * a;
            }
        }

        // reduce across the 16 lanes holding each row, emit per-frame score
#pragma unroll
        for (int v = 0; v < 8; ++v) {
            float nu = num_[v], sq = sq_[v];
#pragma unroll
            for (int msk = 8; msk > 0; msk >>= 1) {
                nu += __shfl_xor(nu, msk, 32);
                sq += __shfl_xor(sq, msk, 32);
            }
            if (ln == 0)
                scores[mbase + 8 * h + v] = nu / (nvrow[v] * sqrtf(sq) + EPSV);
        }
    }
    __syncthreads();

    // ---- logsumexp over the 1024 frame scores ---------------------------
    float lmax = -INFINITY;
    for (int r = tid; r < TV; r += 256) lmax = fmaxf(lmax, scores[r]);
    red[tid] = lmax;
    __syncthreads();
    for (int s = 128; s > 0; s >>= 1) {
        if (tid < s) red[tid] = fmaxf(red[tid], red[tid + s]);
        __syncthreads();
    }
    const float gmax = red[0];
    __syncthreads();
    float lsum = 0.f;
    for (int r = tid; r < TV; r += 256) lsum += __expf(scores[r] - gmax);
    red[tid] = lsum;
    __syncthreads();
    for (int s = 128; s > 0; s >>= 1) {
        if (tid < s) red[tid] += red[tid + s];
        __syncthreads();
    }
    if (tid == 0) out[(size_t)i * BATCH + j] = logf(red[0]) + gmax;
}

// ---------------------------------------------------------------------------
// Host launcher
// ---------------------------------------------------------------------------
extern "C" void kernel_launch(void* const* d_in, const int* in_sizes, int n_in,
                              void* d_out, int out_size, void* d_ws, size_t ws_size,
                              hipStream_t stream) {
    const float* videos    = (const float*)d_in[0];   // [32,1024,1024]
    const float* sentences = (const float*)d_in[1];   // [32,64,1024]
    // d_in[2] = lengths (unused by the reference)
    const float* Wv = (const float*)d_in[3];          // [1024,512]
    const float* bv = (const float*)d_in[4];          // [512]
    const float* Ws = (const float*)d_in[5];          // [1024,512]
    const float* bs = (const float*)d_in[6];          // [512]
    float* out = (float*)d_out;                       // [32,32]

    // Workspace partition:
    //   vrep  bf16 32 MiB | srep bf16 2 MiB | srepT bf16 2 MiB
    //   Wvt   bf16  1 MiB | Wst  bf16 1 MiB | nv f32 128 KiB | ns f32 8 KiB
    char* ws = (char*)d_ws;
    __bf16* vrep  = (__bf16*)ws;                                   ws += (size_t)BATCH * TV * DDIM * 2;
    __bf16* srep  = (__bf16*)ws;                                   ws += (size_t)BATCH * TS * DDIM * 2;
    __bf16* srepT = (__bf16*)ws;                                   ws += (size_t)BATCH * TS * DDIM * 2;
    __bf16* Wvt   = (__bf16*)ws;                                   ws += (size_t)DIN * DDIM * 2;
    __bf16* Wst   = (__bf16*)ws;                                   ws += (size_t)DIN * DDIM * 2;
    float*  nv    = (float*)ws;                                    ws += (size_t)BATCH * TV * 4;
    float*  ns    = (float*)ws;

    // Weight transposes (f32 [K x N] -> bf16 [N x K])
    transpose_w_bf16<<<(DIN * DDIM + 255) / 256, 256, 0, stream>>>(Wv, Wvt, DIN, DDIM);
    transpose_w_bf16<<<(DIN * DDIM + 255) / 256, 256, 0, stream>>>(Ws, Wst, DIN, DDIM);

    // Encoders (M x 1024 x 512 GEMMs, bf16 WMMA)
    encode_gemm<<<dim3(DDIM / 64, (BATCH * TV) / 64), 256, 0, stream>>>(
        videos, Wvt, bv, vrep, BATCH * TV, DIN, DDIM);
    encode_gemm<<<dim3(DDIM / 64, (BATCH * TS) / 64), 256, 0, stream>>>(
        sentences, Wst, bs, srep, BATCH * TS, DIN, DDIM);

    // srep -> srep^T (feeds the TDM descriptor's contiguous 2D tile)
    transpose_srep<<<(BATCH * TS * DDIM) / 256, 256, 0, stream>>>(srep, srepT);

    // Row norms
    row_norms<<<(BATCH * TV) / 8, 256, 0, stream>>>(vrep, nv, BATCH * TV);
    row_norms<<<(BATCH * TS) / 8, 256, 0, stream>>>(srep, ns, BATCH * TS);

    // All-pairs co-attention scores
    pair_kernel<<<dim3(BATCH, BATCH), 256, SMEM_PAIR, stream>>>(
        vrep, srep, srepT, nv, ns, out);
}
